// MultiHeadAttention_42021960024190
// MI455X (gfx1250) — matrix operand order repaired and checked by reading, not compiled, not run on hardware
//
#include <hip/hip_runtime.h>
#include <hip/hip_bf16.h>
#include <math.h>

// ---------------- problem constants ----------------
constexpr int kE = 1024;          // embed dim
constexpr int kH = 16;            // heads
constexpr int kD = 64;            // head dim
constexpr int kS = 2048;          // sequence
constexpr int kB = 2;             // batch
constexpr int kRows = kS * kB;    // 4096 rows of the (S,B,E) tensors

typedef __attribute__((ext_vector_type(16))) __bf16 bf16x16;
typedef __attribute__((ext_vector_type(8)))  float  floatx8;

// D = A(16x32 bf16) * B(32x16 bf16) + C(16x16 f32)
static __device__ inline floatx8 wmma_bf16(bf16x16 a, bf16x16 b, floatx8 c) {
    return __builtin_amdgcn_wmma_f32_16x16x32_bf16(
        /*neg_a=*/false, a, /*neg_b=*/false, b,
        /*c_mod=*/(short)0, c, /*reuse_a=*/false, /*reuse_b=*/false);
}

// A fragment: 16(M) x 32(K), p points at element [m=0,k=0], row stride ld.
// ISA 16-bit A layout: lanes 0-15 hold K=0..7,16..23; lanes 16-31 hold K=8..15,24..31.
static __device__ inline bf16x16 frag_a(const __bf16* p, int ld) {
    const int lane = threadIdx.x & 31;
    const int m  = lane & 15;
    const int hi = lane >> 4;
    bf16x16 f;
#pragma unroll
    for (int v = 0; v < 8; ++v) {
        const int k = ((v & 4) ? 16 : 0) + hi * 8 + (v & 3) * 2;
        f[2 * v]     = p[m * ld + k];
        f[2 * v + 1] = p[m * ld + k + 1];
    }
    return f;
}

// B fragment: 32(K) x 16(N), stored TRANSPOSED in memory as p[n*ld + k]
// ISA 16-bit B layout: lanes 0-15 hold K=0..7,16..23; lanes 16-31 hold K=8..15,24..31.
static __device__ inline bf16x16 frag_b(const __bf16* p, int ld) {
    const int lane = threadIdx.x & 31;
    const int n  = lane & 15;
    const int hi = lane >> 4;
    bf16x16 f;
#pragma unroll
    for (int v = 0; v < 8; ++v) {
        const int k = ((v & 4) ? 16 : 0) + hi * 8 + (v & 3) * 2;
        f[2 * v]     = p[n * ld + k];
        f[2 * v + 1] = p[n * ld + k + 1];
    }
    return f;
}

// ============================================================================
// Kernel 1: fused Q/K/V projection.  Y = X @ W^T + b, downconvert fp32->bf16.
// blockIdx.z = 0:Q 1:K 2:V.  Block = 128 threads (4 waves), 64x64 output tile.
// q/k stored [h][b][s][d] bf16; v stored transposed [h][b][d][s] bf16 so the
// later P@V B-operand is K-pair-contiguous.
// ============================================================================
__global__ __launch_bounds__(128)
void qkv_proj_kernel(const float* __restrict__ q_in,
                     const float* __restrict__ k_in,
                     const float* __restrict__ v_in,
                     const float* __restrict__ Wq, const float* __restrict__ bq,
                     const float* __restrict__ Wk, const float* __restrict__ bk,
                     const float* __restrict__ Wv, const float* __restrict__ bv,
                     __bf16* __restrict__ q_ws,
                     __bf16* __restrict__ k_ws,
                     __bf16* __restrict__ vt_ws) {
    __shared__ __bf16 at[64][32];   // X rows (M x K)
    __shared__ __bf16 bt[64][32];   // W rows = output cols (N x K)

    const int g = blockIdx.z;
    const float* X    = (g == 0) ? q_in : (g == 1) ? k_in : v_in;
    const float* W    = (g == 0) ? Wq   : (g == 1) ? Wk   : Wv;
    const float* bias = (g == 0) ? bq   : (g == 1) ? bk   : bv;

    const int row0 = blockIdx.y * 64;
    const int col0 = blockIdx.x * 64;
    const int tid  = threadIdx.x;
    const int wave = tid >> 5;

    floatx8 acc[4] = {};

    for (int k0 = 0; k0 < kE; k0 += 32) {
        // cooperative fp32 load + bf16 downconvert into LDS
        const int r  = tid >> 1;
        const int h4 = (tid & 1) * 16;
        const float* xs = X + (size_t)(row0 + r) * kE + k0 + h4;
        const float* ws = W + (size_t)(col0 + r) * kE + k0 + h4;
        if (k0 + 32 < kE) {  // prefetch next K-slab (-> global_prefetch_b8)
            __builtin_prefetch(xs + 32, 0, 1);
            __builtin_prefetch(ws + 32, 0, 1);
        }
#pragma unroll
        for (int j = 0; j < 16; j += 4) {
            float4 xv = *(const float4*)(xs + j);
            at[r][h4 + j + 0] = (__bf16)xv.x;
            at[r][h4 + j + 1] = (__bf16)xv.y;
            at[r][h4 + j + 2] = (__bf16)xv.z;
            at[r][h4 + j + 3] = (__bf16)xv.w;
            float4 wv = *(const float4*)(ws + j);
            bt[r][h4 + j + 0] = (__bf16)wv.x;
            bt[r][h4 + j + 1] = (__bf16)wv.y;
            bt[r][h4 + j + 2] = (__bf16)wv.z;
            bt[r][h4 + j + 3] = (__bf16)wv.w;
        }
        __syncthreads();

        bf16x16 a = frag_a(&at[wave * 16][0], 32);
#pragma unroll
        for (int ct = 0; ct < 4; ++ct) {
            bf16x16 b = frag_b(&bt[ct * 16][0], 32);
            acc[ct] = wmma_bf16(a, b, acc[ct]);
        }
        __syncthreads();
    }

    // epilogue: C/D layout -> scatter into head-split bf16 workspace
    const int lane = tid & 31;
    const int n  = lane & 15;
    const int hi = lane >> 4;
#pragma unroll
    for (int ct = 0; ct < 4; ++ct) {
#pragma unroll
        for (int r = 0; r < 8; ++r) {
            const int m   = r + hi * 8;
            const int row = row0 + wave * 16 + m;   // row = s*B + b
            const int col = col0 + ct * 16 + n;     // col = h*64 + d
            const float val = acc[ct][r] + bias[col];
            const int s  = row / kB;
            const int bb = row % kB;
            const int h  = col >> 6;
            const int d  = col & 63;
            if (g == 2) {
                vt_ws[((size_t)(h * kB + bb) * kD + d) * kS + s] = (__bf16)val;
            } else {
                __bf16* dst = (g == 0) ? q_ws : k_ws;
                dst[((size_t)(h * kB + bb) * kS + s) * kD + d] = (__bf16)val;
            }
        }
    }
}

// ============================================================================
// Kernel 2: attention for one (h, b, 16-row q tile). 256 threads = 8 waves.
// Dynamic LDS: f32 scores 16x2048 (128KB) + bf16 probs 16x2048 (64KB) +
// reductions + out accumulator (~5KB) = ~198KB (CDNA5: 320KB/WGP).
// ============================================================================
__global__ __launch_bounds__(256)
void attention_kernel(const __bf16* __restrict__ q_ws,
                      const __bf16* __restrict__ k_ws,
                      const __bf16* __restrict__ vt_ws,
                      const float* __restrict__ attn_mask,
                      const unsigned char* __restrict__ pad_mask,
                      __bf16* __restrict__ ctx,
                      float* __restrict__ attn_out) {
    extern __shared__ char smem[];
    float*  sc      = (float*)smem;                                  // [16][kS]
    __bf16* pb      = (__bf16*)(smem + 16 * kS * 4);                 // [16][kS]
    float*  red     = (float*)(smem + 16 * kS * 4 + 16 * kS * 2);    // [16][16]
    float*  rowstat = red + 256;                                     // [16]
    float*  outacc  = rowstat + 16;                                  // [16][64]

    const int q0 = blockIdx.x * 16;
    const int hb = blockIdx.y;                 // h*kB + b
    const int bb = hb % kB;
    const int h  = hb / kB;
    const int tid  = threadIdx.x;
    const int wave = tid >> 5;
    const int lane = tid & 31;
    const int n_l  = lane & 15;
    const int hi   = lane >> 4;

    const __bf16* qg = q_ws  + ((size_t)hb * kS + q0) * kD;  // 16 x 64, ld=64
    const __bf16* kg = k_ws  + (size_t)hb * kS * kD;         // 2048 x 64, ld=64
    const __bf16* vg = vt_ws + (size_t)hb * kD * kS;         // 64 x 2048, ld=2048
    const float scale = 0.125f;                              // 1/sqrt(64)

    for (int i = tid; i < 16 * 64; i += 256) outacc[i] = 0.0f;

    // -------- scores: S = scale * Q K^T + mask ; each wave owns 256 cols ----
    // Q fragments are loop-invariant: keep both 16x32 A-fragments in VGPRs.
    bf16x16 aq0 = frag_a(qg, kD);
    bf16x16 aq1 = frag_a(qg + 32, kD);
    for (int ct = 0; ct < 16; ++ct) {
        const int n0 = wave * 256 + ct * 16;
        if (ct + 1 < 16) {  // prefetch next 16-key tile (2KB, sequential)
            __builtin_prefetch(kg + (size_t)(n0 + 16) * kD + (lane << 2), 0, 1);
        }
        floatx8 acc = {};
        bf16x16 b0 = frag_b(kg + (size_t)n0 * kD, kD);
        acc = wmma_bf16(aq0, b0, acc);
        bf16x16 b1 = frag_b(kg + (size_t)n0 * kD + 32, kD);
        acc = wmma_bf16(aq1, b1, acc);
#pragma unroll
        for (int r = 0; r < 8; ++r) {
            const int m   = r + hi * 8;
            const int col = n0 + n_l;
            float v = acc[r] * scale + attn_mask[(size_t)(q0 + m) * kS + col];
            if (pad_mask[bb * kS + col]) v = -INFINITY;
            sc[m * kS + col] = v;
        }
    }
    __syncthreads();

    // -------- softmax over 2048 cols: row = tid>>4, 128-elem segments -------
    {
        const int row = tid >> 4, seg = tid & 15;
        float m0 = -INFINITY;
        for (int i = seg * 128; i < seg * 128 + 128; ++i)
            m0 = fmaxf(m0, sc[row * kS + i]);
        red[row * 16 + seg] = m0;
    }
    __syncthreads();
    if ((tid & 15) == 0) {
        const int row = tid >> 4;
        float m0 = red[row * 16];
        for (int i = 1; i < 16; ++i) m0 = fmaxf(m0, red[row * 16 + i]);
        rowstat[row] = m0;
    }
    __syncthreads();
    {
        const int row = tid >> 4, seg = tid & 15;
        const float rm = rowstat[row];
        float sum = 0.0f;
        for (int i = seg * 128; i < seg * 128 + 128; ++i) {
            float e = __expf(sc[row * kS + i] - rm);
            sc[row * kS + i] = e;
            sum += e;
        }
        red[row * 16 + seg] = sum;
    }
    __syncthreads();
    if ((tid & 15) == 0) {
        const int row = tid >> 4;
        float s0 = 0.0f;
        for (int i = 0; i < 16; ++i) s0 += red[row * 16 + i];
        rowstat[row] = 1.0f / s0;
    }
    __syncthreads();
    // -------- normalize: stream fp32 weights to HBM once, keep bf16 in LDS --
    {
        const int row = tid >> 4, seg = tid & 15;
        const float rinv = rowstat[row];
        float* ao = attn_out + ((size_t)(bb * kH + h) * kS + (q0 + row)) * kS;
        for (int i = seg * 128; i < seg * 128 + 128; ++i) {
            const float p = sc[row * kS + i] * rinv;
            ao[i] = p;
            pb[row * kS + i] = (__bf16)p;
        }
    }
    __syncthreads();

    // -------- P @ V : each wave reduces its 256-key chunk, ds_add combine ---
    floatx8 acc2[4] = {};
    for (int ks = 0; ks < 8; ++ks) {
        const int k0 = wave * 256 + ks * 32;
        if (ks + 1 < 8) {  // prefetch next V chunk rows
            __builtin_prefetch(vg + (size_t)(lane & 3) * 16 * kS + k0 + 32, 0, 1);
        }
        bf16x16 a = frag_a(pb + k0, kS);
#pragma unroll
        for (int dt = 0; dt < 4; ++dt) {
            bf16x16 b = frag_b(vg + (size_t)(dt * 16) * kS + k0, kS);
            acc2[dt] = wmma_bf16(a, b, acc2[dt]);
        }
    }
#pragma unroll
    for (int dt = 0; dt < 4; ++dt) {
#pragma unroll
        for (int r = 0; r < 8; ++r) {
            const int m = r + hi * 8;
            atomicAdd(&outacc[m * 64 + dt * 16 + n_l], acc2[dt][r]);
        }
    }
    __syncthreads();

    // -------- write context tile (row = s*B+b, col = h*64+d) as bf16 --------
    for (int i = tid; i < 16 * 64; i += 256) {
        const int m = i >> 6, d = i & 63;
        ctx[((size_t)(q0 + m) * kB + bb) * kE + h * kD + d] = (__bf16)outacc[i];
    }
}

// ============================================================================
// Kernel 3: output projection  out = ctx @ Wo^T + bo  (fp32 out)
// ============================================================================
__global__ __launch_bounds__(128)
void out_proj_kernel(const __bf16* __restrict__ ctx,
                     const float* __restrict__ Wo,
                     const float* __restrict__ bo,
                     float* __restrict__ out) {
    __shared__ __bf16 at[64][32];
    __shared__ __bf16 bt[64][32];

    const int row0 = blockIdx.y * 64;
    const int col0 = blockIdx.x * 64;
    const int tid  = threadIdx.x;
    const int wave = tid >> 5;

    floatx8 acc[4] = {};

    for (int k0 = 0; k0 < kE; k0 += 32) {
        const int r  = tid >> 1;
        const int h4 = (tid & 1) * 16;
        const __bf16* cs = ctx + (size_t)(row0 + r) * kE + k0 + h4;
        const float*  ws = Wo  + (size_t)(col0 + r) * kE + k0 + h4;
        if (k0 + 32 < kE) {
            __builtin_prefetch(cs + 32, 0, 1);
            __builtin_prefetch(ws + 32, 0, 1);
        }
#pragma unroll
        for (int j = 0; j < 16; ++j) at[r][h4 + j] = cs[j];
#pragma unroll
        for (int j = 0; j < 16; j += 4) {
            float4 wv = *(const float4*)(ws + j);
            bt[r][h4 + j + 0] = (__bf16)wv.x;
            bt[r][h4 + j + 1] = (__bf16)wv.y;
            bt[r][h4 + j + 2] = (__bf16)wv.z;
            bt[r][h4 + j + 3] = (__bf16)wv.w;
        }
        __syncthreads();

        bf16x16 a = frag_a(&at[wave * 16][0], 32);
#pragma unroll
        for (int ct = 0; ct < 4; ++ct) {
            bf16x16 b = frag_b(&bt[ct * 16][0], 32);
            acc[ct] = wmma_bf16(a, b, acc[ct]);
        }
        __syncthreads();
    }

    const int lane = tid & 31;
    const int n  = lane & 15;
    const int hi = lane >> 4;
#pragma unroll
    for (int ct = 0; ct < 4; ++ct) {
#pragma unroll
        for (int r = 0; r < 8; ++r) {
            const int m   = r + hi * 8;
            const int row = row0 + wave * 16 + m;
            const int col = col0 + ct * 16 + n;
            out[(size_t)row * kE + col] = acc[ct][r] + bo[col];
        }
    }
}

// ============================================================================
// launcher
// ============================================================================
extern "C" void kernel_launch(void* const* d_in, const int* in_sizes, int n_in,
                              void* d_out, int out_size, void* d_ws, size_t ws_size,
                              hipStream_t stream) {
    (void)in_sizes; (void)n_in; (void)out_size; (void)ws_size;

    const float* query     = (const float*)d_in[0];
    const float* key_      = (const float*)d_in[1];
    const float* value     = (const float*)d_in[2];
    const float* attn_mask = (const float*)d_in[3];
    const unsigned char* pad_mask = (const unsigned char*)d_in[4];
    const float* Wq = (const float*)d_in[5];
    const float* bq = (const float*)d_in[6];
    const float* Wk = (const float*)d_in[7];
    const float* bk = (const float*)d_in[8];
    const float* Wv = (const float*)d_in[9];
    const float* bv = (const float*)d_in[10];
    const float* Wo = (const float*)d_in[11];
    const float* bo = (const float*)d_in[12];

    // workspace: q(8MB) | k(8MB) | v^T(8MB) | ctx(8MB), all bf16
    char* ws = (char*)d_ws;
    const size_t seg = (size_t)kRows * kE * sizeof(__bf16);
    __bf16* q_ws  = (__bf16*)(ws);
    __bf16* k_ws  = (__bf16*)(ws + seg);
    __bf16* vt_ws = (__bf16*)(ws + 2 * seg);
    __bf16* ctx   = (__bf16*)(ws + 3 * seg);

    float* out  = (float*)d_out;                  // (S,B,E) fp32
    float* attn = out + (size_t)kRows * kE;       // (B,H,S,S) fp32

    // 1) Q/K/V projections
    qkv_proj_kernel<<<dim3(kE / 64, kRows / 64, 3), 128, 0, stream>>>(
        query, key_, value, Wq, bq, Wk, bk, Wv, bv, q_ws, k_ws, vt_ws);

    // 2) fused scores + softmax + weight writeout + P@V
    const size_t smem_bytes = (size_t)16 * kS * 4   // f32 scores
                            + (size_t)16 * kS * 2   // bf16 probs
                            + (256 + 16 + 16 * 64) * 4;
    attention_kernel<<<dim3(kS / 16, kH * kB), 256, smem_bytes, stream>>>(
        q_ws, k_ws, vt_ws, attn_mask, pad_mask, ctx, attn);

    // 3) output projection
    out_proj_kernel<<<dim3(kE / 64, kRows / 64), 128, 0, stream>>>(
        ctx, Wo, bo, out);
}
